// TSTEncoder_48481590837767
// MI455X (gfx1250) — matrix operand order repaired and checked by reading, not compiled, not run on hardware
//
#include <hip/hip_runtime.h>
#include <hip/hip_bf16.h>
#include <math.h>

// ---------------------------------------------------------------------------
// TST encoder block for MI455X (gfx1250, wave32).
// All GEMMs via v_wmma_f32_16x16x32_f16 (f16 in, f32 accumulate).
// Attention fused flash-style: prev_attn (512MB) streamed exactly once with
// non-temporal loads; tile staging uses GLOBAL_LOAD_ASYNC_TO_LDS (ASYNCcnt)
// via the gfx1250 builtins.
// ---------------------------------------------------------------------------

typedef __attribute__((ext_vector_type(16))) _Float16 v16h;
typedef __attribute__((ext_vector_type(8)))  float    v8f;
typedef __attribute__((ext_vector_type(4)))  int      i32x4;

#define BB 4
#define SS 2048
#define DD 256
#define HH 8
#define DH 32
#define FFN 1024
#define ROWS (BB * SS)   // 8192

#if defined(__has_builtin)
#  if __has_builtin(__builtin_amdgcn_global_load_async_to_lds_b128)
#    define HAVE_ASYNC_LDS 1
#  endif
#  if __has_builtin(__builtin_amdgcn_s_wait_asynccnt)
#    define HAVE_WAIT_ASYNC 1
#  endif
#endif

#if defined(HAVE_ASYNC_LDS)
#define AS1 __attribute__((address_space(1)))
#define AS3 __attribute__((address_space(3)))
__device__ __forceinline__ void async_b128(const void* g, void* l) {
  // builtin signature (from toolchain diagnostic): (i32x4 AS1*, i32x4 AS3*, Ii, Ii)
  __builtin_amdgcn_global_load_async_to_lds_b128(
      (AS1 i32x4*)g, (AS3 i32x4*)l, 0, 0);
}
__device__ __forceinline__ void wait_async0() {
#if defined(HAVE_WAIT_ASYNC)
  __builtin_amdgcn_s_wait_asynccnt(0);
#else
  asm volatile("s_wait_asynccnt 0x0" ::: "memory");
#endif
}
#endif

__device__ __forceinline__ v8f wmma_f16(v16h a, v16h b, v8f c) {
  // (neg_a, A, neg_b, B, c_mod, C, reuse_a, reuse_b)
  return __builtin_amdgcn_wmma_f32_16x16x32_f16(false, a, false, b, (short)0, c,
                                                false, false);
}

// A fragment (16x32, f16): lane = M (mod 16); K split by lane-half.
// VGPR j : K = 16*half + 2j
__device__ __forceinline__ v16h load_a_frag(const _Float16* A, int lda) {
  const int l = threadIdx.x & 31;
  const int half = l >> 4, m = l & 15;
  v16h a;
#pragma unroll
  for (int j = 0; j < 8; ++j) {
    const int k = 16 * half + 2 * j;
    a[2 * j]     = A[m * lda + k];
    a[2 * j + 1] = A[m * lda + k + 1];
  }
  return a;
}

// B fragment (32x16, f16): lane = N; K rows 2-per-VGPR, lane-half selects K 0..15 / 16..31.
__device__ __forceinline__ v16h load_b_frag(const _Float16* B, int ldb) {
  const int l = threadIdx.x & 31;
  const int kg = (l >> 4) * 16, n = l & 15;
  v16h b;
#pragma unroll
  for (int j = 0; j < 8; ++j) {
    b[2 * j]     = B[(kg + 2 * j) * ldb + n];
    b[2 * j + 1] = B[(kg + 2 * j + 1) * ldb + n];
  }
  return b;
}

// ---------------------------------------------------------------------------
// f32 -> f16 cast (weights)
// ---------------------------------------------------------------------------
__global__ void cast_f16_kernel(const float* __restrict__ in,
                                _Float16* __restrict__ out, int n) {
  int i = blockIdx.x * 256 + threadIdx.x;
  if (i < n) out[i] = (_Float16)in[i];
}

// ---------------------------------------------------------------------------
// LayerNorm over D=256; one row per 256-thread block. Emits f32 + f16 copies.
// ---------------------------------------------------------------------------
__global__ __launch_bounds__(256) void ln_kernel(const float* __restrict__ x,
                                                 const float* __restrict__ g,
                                                 const float* __restrict__ bta,
                                                 float* __restrict__ y32,
                                                 _Float16* __restrict__ y16) {
  __shared__ float red[256];
  const int row = blockIdx.x;
  const int t = threadIdx.x;
  const float v = x[(size_t)row * DD + t];

  red[t] = v;
  __syncthreads();
  for (int s = 128; s > 0; s >>= 1) {
    if (t < s) red[t] += red[t + s];
    __syncthreads();
  }
  const float mu = red[0] * (1.0f / DD);
  __syncthreads();

  const float d = v - mu;
  red[t] = d * d;
  __syncthreads();
  for (int s = 128; s > 0; s >>= 1) {
    if (t < s) red[t] += red[t + s];
    __syncthreads();
  }
  const float var = red[0] * (1.0f / DD);

  const float nrm = d * rsqrtf(var + 1e-5f) * g[t] + bta[t];
  y32[(size_t)row * DD + t] = nrm;
  y16[(size_t)row * DD + t] = (_Float16)nrm;
}

// ---------------------------------------------------------------------------
// Generic WMMA GEMM: C[M,N] = A[M,K] * Bw[K,N] (+ epilogue per MODE)
//  MODE 0: QKV -> f16 out in [B,H,S,DH] head layout (+bias)
//  MODE 1: O-proj -> f32 out = acc + bias + resid (xn residual)
//  MODE 2: FFN1 -> f16 out = gelu_exact(acc + bias)
//  MODE 3: FFN2 -> f32 out = acc + bias + resid (x2 residual)  == d_out (NT)
// Block: 256 threads = 8 waves, 64x64 tile, K step 32.
// LDS row pitches (80B / 144B) are multiples of 16B -> aligned B128 async.
// ---------------------------------------------------------------------------
template <int MODE>
__global__ __launch_bounds__(256) void gemm_wmma_kernel(
    const _Float16* __restrict__ A, const _Float16* __restrict__ Bw,
    const float* __restrict__ bias, const float* __restrict__ resid,
    void* __restrict__ out, int M, int N, int K) {
  __shared__ _Float16 As[64][40];   // 64x32 tile (+pad), pitch 80B
  __shared__ _Float16 Bs[32][72];   // 32x64 tile (+pad), pitch 144B

  const int wid = threadIdx.x >> 5;
  const int wm = wid & 3;           // 0..3 -> 16-row slab
  const int wn = wid >> 2;          // 0..1 -> 32-col slab
  const int row0 = blockIdx.y * 64;
  const int col0 = blockIdx.x * 64;

  v8f acc0 = {}; v8f acc1 = {};

  for (int k0 = 0; k0 < K; k0 += 32) {
#if defined(HAVE_ASYNC_LDS)
    {
      // A tile: 64 rows x 4 b128 segments; B tile: 32 rows x 8 b128 segments.
      const int t = threadIdx.x;
      const int ar = t >> 2, aseg = t & 3;
      async_b128(A + (size_t)(row0 + ar) * K + k0 + aseg * 8,
                 &As[ar][aseg * 8]);
      const int br = t >> 3, bseg = t & 7;
      async_b128(Bw + (size_t)(k0 + br) * N + col0 + bseg * 8,
                 &Bs[br][bseg * 8]);
      wait_async0();
    }
#else
    for (int t = threadIdx.x; t < 64 * 16; t += 256) {   // A: 64x32 halves
      const int r = t >> 4, c2 = (t & 15) * 2;
      As[r][c2]     = A[(size_t)(row0 + r) * K + k0 + c2];
      As[r][c2 + 1] = A[(size_t)(row0 + r) * K + k0 + c2 + 1];
    }
    for (int t = threadIdx.x; t < 32 * 32; t += 256) {   // B: 32x64 halves
      const int r = t >> 5, c2 = (t & 31) * 2;
      Bs[r][c2]     = Bw[(size_t)(k0 + r) * N + col0 + c2];
      Bs[r][c2 + 1] = Bw[(size_t)(k0 + r) * N + col0 + c2 + 1];
    }
#endif
    __syncthreads();

    const v16h a  = load_a_frag(&As[wm * 16][0], 40);
    const v16h b0 = load_b_frag(&Bs[0][wn * 32], 72);
    const v16h b1 = load_b_frag(&Bs[0][wn * 32 + 16], 72);
    acc0 = wmma_f16(a, b0, acc0);
    acc1 = wmma_f16(a, b1, acc1);
    __syncthreads();
  }

  // C layout: element (M = r + 8*(lane/16), N = lane%16)
  const int l = threadIdx.x & 31;
  const int half = l >> 4, n = l & 15;
#pragma unroll
  for (int r = 0; r < 8; ++r) {
    const int m = r + 8 * half;
    const int grow = row0 + wm * 16 + m;
#pragma unroll
    for (int t = 0; t < 2; ++t) {
      const int gcol = col0 + wn * 32 + t * 16 + n;
      float v = (t ? acc1[r] : acc0[r]) + bias[gcol];
      if constexpr (MODE == 0) {
        // scatter to [B,H,S,DH] f16
        const int bb = grow >> 11, s = grow & (SS - 1);
        const int h = gcol >> 5, dh = gcol & 31;
        ((_Float16*)out)[(((size_t)(bb * HH + h) * SS + s) * DH) + dh] =
            (_Float16)v;
      } else if constexpr (MODE == 1) {
        v += resid[(size_t)grow * DD + gcol];
        ((float*)out)[(size_t)grow * DD + gcol] = v;
      } else if constexpr (MODE == 2) {
        const float gl = 0.5f * v * (1.0f + erff(v * 0.70710678118654752f));
        ((_Float16*)out)[(size_t)grow * FFN + gcol] = (_Float16)gl;
      } else {
        v += resid[(size_t)grow * DD + gcol];
        __builtin_nontemporal_store(v, &((float*)out)[(size_t)grow * DD + gcol]);
      }
    }
  }
}

// ---------------------------------------------------------------------------
// Fused flash attention with prev_attn bias.
// grid = (S/128, B*H); block = 256 (8 waves); each wave owns 16 query rows.
// K/V chunks staged via async-to-LDS; prev_attn read non-temporally.
// ---------------------------------------------------------------------------
__global__ __launch_bounds__(256) void attn_kernel(
    const _Float16* __restrict__ q, const _Float16* __restrict__ k,
    const _Float16* __restrict__ v, const float* __restrict__ prev,
    const float* __restrict__ scale_p, _Float16* __restrict__ ctx_out) {
  __shared__ _Float16 kbuf[32][40];        // 32 keys x DH, pitch 80B
  __shared__ _Float16 vbuf[32][40];        // 32 keys x DH, pitch 80B
  __shared__ _Float16 pbuf[8][16][40];     // per-wave P tile 16x32

  const int bh = blockIdx.y;               // 0..31
  const int b = bh >> 3, h = bh & 7;
  const int wid = threadIdx.x >> 5;
  const int q0 = blockIdx.x * 128 + wid * 16;
  const float scl = scale_p[0];

  const _Float16* qh = q + (size_t)bh * SS * DH;
  const _Float16* kh = k + (size_t)bh * SS * DH;
  const _Float16* vh = v + (size_t)bh * SS * DH;
  const float*    ph = prev + (size_t)bh * SS * SS;

  // Q fragment loaded once straight from global (A layout, lda = DH)
  const v16h aq = load_a_frag(qh + (size_t)q0 * DH, DH);

  const int l = threadIdx.x & 31;
  const int half = l >> 4, n = l & 15;

  float mrow[8], lrow[8];
  v8f acc0 = {}; v8f acc1 = {};
#pragma unroll
  for (int r = 0; r < 8; ++r) { mrow[r] = -1e30f; lrow[r] = 0.0f; }

  for (int kc = 0; kc < SS; kc += 32) {
#if defined(HAVE_ASYNC_LDS)
    {
      // 32 keys x 32dh f16 = 128 b128 segments per buffer; waves 0-3 -> K,
      // waves 4-7 -> V (uniform per wave).
      const int t = threadIdx.x;
      const int idx = t & 127;
      const int rr = idx >> 2, seg = idx & 3;
      const _Float16* src = (t < 128) ? kh : vh;
      _Float16* dst = (t < 128) ? &kbuf[rr][seg * 8] : &vbuf[rr][seg * 8];
      async_b128(src + (size_t)(kc + rr) * DH + seg * 8, dst);
      wait_async0();
    }
#else
    for (int t = threadIdx.x; t < 32 * 32; t += 256) {
      const int rr = t >> 5, cc = t & 31;
      kbuf[rr][cc] = kh[(size_t)(kc + rr) * DH + cc];
      vbuf[rr][cc] = vh[(size_t)(kc + rr) * DH + cc];
    }
#endif
    // prefetch next prev_attn chunk (global_prefetch_b8)
    if (kc + 32 < SS)
      __builtin_prefetch(ph + (size_t)(q0 + (l & 15)) * SS + kc + 32, 0, 1);
    __syncthreads();

    // scores: A = Q (16 x 32dh), B = K^T tiles (32dh x 16 keys)
    v16h bk0, bk1;
#pragma unroll
    for (int j = 0; j < 8; ++j) {
      const int kk = half * 16 + 2 * j;     // dh row pair within K^T
      bk0[2 * j]     = kbuf[n][kk];
      bk0[2 * j + 1] = kbuf[n][kk + 1];
      bk1[2 * j]     = kbuf[16 + n][kk];
      bk1[2 * j + 1] = kbuf[16 + n][kk + 1];
    }
    v8f s0 = {}, s1 = {};
    s0 = wmma_f16(aq, bk0, s0);
    s1 = wmma_f16(aq, bk1, s1);

    // fuse scale + prev_attn bias (non-temporal: single-use 512MB stream),
    // then online softmax per row
#pragma unroll
    for (int r = 0; r < 8; ++r) {
      const int m = r + 8 * half;
      const size_t base = (size_t)(q0 + m) * SS + kc;
      s0[r] = s0[r] * scl + __builtin_nontemporal_load(&ph[base + n]);
      s1[r] = s1[r] * scl + __builtin_nontemporal_load(&ph[base + 16 + n]);

      float mx = fmaxf(s0[r], s1[r]);
#pragma unroll
      for (int off = 1; off < 16; off <<= 1)
        mx = fmaxf(mx, __shfl_xor(mx, off, 32));
      const float mnew = fmaxf(mrow[r], mx);
      const float alpha = __expf(mrow[r] - mnew);
      const float p0 = __expf(s0[r] - mnew);
      const float p1 = __expf(s1[r] - mnew);
      float ps = p0 + p1;
#pragma unroll
      for (int off = 1; off < 16; off <<= 1) ps += __shfl_xor(ps, off, 32);
      lrow[r] = lrow[r] * alpha + ps;
      mrow[r] = mnew;
      acc0[r] *= alpha;
      acc1[r] *= alpha;
      pbuf[wid][m][n]      = (_Float16)p0;
      pbuf[wid][m][16 + n] = (_Float16)p1;
    }
    __syncthreads();

    // ctx += P (16x32 keys) * V (32 keys x 32 dh)
    const v16h ap = load_a_frag(&pbuf[wid][0][0], 40);
    v16h bv0, bv1;
#pragma unroll
    for (int j = 0; j < 8; ++j) {
      const int kk = half * 16 + 2 * j;     // key row pair within chunk
      bv0[2 * j]     = vbuf[kk][n];
      bv0[2 * j + 1] = vbuf[kk + 1][n];
      bv1[2 * j]     = vbuf[kk][16 + n];
      bv1[2 * j + 1] = vbuf[kk + 1][16 + n];
    }
    acc0 = wmma_f16(ap, bv0, acc0);
    acc1 = wmma_f16(ap, bv1, acc1);
    __syncthreads();
  }

  // normalize and write ctx as f16 in [B,S,D] (col = h*32 + dh)
#pragma unroll
  for (int r = 0; r < 8; ++r) {
    const int m = r + 8 * half;
    const float inv = 1.0f / lrow[r];
    const size_t rowbase = ((size_t)b * SS + (q0 + m)) * DD + h * DH;
    ctx_out[rowbase + n]      = (_Float16)(acc0[r] * inv);
    ctx_out[rowbase + 16 + n] = (_Float16)(acc1[r] * inv);
  }
}

// ---------------------------------------------------------------------------
// Host-side orchestration
// ---------------------------------------------------------------------------
extern "C" void kernel_launch(void* const* d_in, const int* in_sizes, int n_in,
                              void* d_out, int out_size, void* d_ws,
                              size_t ws_size, hipStream_t stream) {
  const float* x        = (const float*)d_in[0];
  const float* prev     = (const float*)d_in[1];
  const float* qw       = (const float*)d_in[2];
  const float* qb       = (const float*)d_in[3];
  const float* kw       = (const float*)d_in[4];
  const float* kb       = (const float*)d_in[5];
  const float* vw       = (const float*)d_in[6];
  const float* vb       = (const float*)d_in[7];
  const float* ow       = (const float*)d_in[8];
  const float* ob       = (const float*)d_in[9];
  const float* scale    = (const float*)d_in[10];
  const float* ln1_g    = (const float*)d_in[11];
  const float* ln1_b    = (const float*)d_in[12];
  const float* ln2_g    = (const float*)d_in[13];
  const float* ln2_b    = (const float*)d_in[14];
  const float* w1       = (const float*)d_in[15];
  const float* b1       = (const float*)d_in[16];
  const float* w2       = (const float*)d_in[17];
  const float* b2       = (const float*)d_in[18];
  float* out = (float*)d_out;

  char* ws = (char*)d_ws;
  size_t off = 0;
  auto alloc = [&](size_t bytes) { char* p = ws + off; off += bytes; return p; };
  _Float16* wQ  = (_Float16*)alloc((size_t)DD * DD * 2);
  _Float16* wK  = (_Float16*)alloc((size_t)DD * DD * 2);
  _Float16* wV  = (_Float16*)alloc((size_t)DD * DD * 2);
  _Float16* wO  = (_Float16*)alloc((size_t)DD * DD * 2);
  _Float16* w1h = (_Float16*)alloc((size_t)DD * FFN * 2);
  _Float16* w2h = (_Float16*)alloc((size_t)FFN * DD * 2);
  float*    xn32   = (float*)alloc((size_t)ROWS * DD * 4);
  _Float16* xn16   = (_Float16*)alloc((size_t)ROWS * DD * 2);
  _Float16* qws    = (_Float16*)alloc((size_t)ROWS * DD * 2);   // [B,H,S,DH]
  _Float16* kws    = (_Float16*)alloc((size_t)ROWS * DD * 2);
  _Float16* vws    = (_Float16*)alloc((size_t)ROWS * DD * 2);
  _Float16* ctx16  = (_Float16*)alloc((size_t)ROWS * DD * 2);   // [B,S,D]
  float*    xatt32 = (float*)alloc((size_t)ROWS * DD * 4);
  float*    x2_32  = (float*)alloc((size_t)ROWS * DD * 4);
  _Float16* x2_16  = (_Float16*)alloc((size_t)ROWS * DD * 2);
  _Float16* hid16  = (_Float16*)alloc((size_t)ROWS * FFN * 2);

  // 1. weight casts f32 -> f16
  const int ndd = DD * DD, nff = DD * FFN;
  cast_f16_kernel<<<(ndd + 255) / 256, 256, 0, stream>>>(qw, wQ, ndd);
  cast_f16_kernel<<<(ndd + 255) / 256, 256, 0, stream>>>(kw, wK, ndd);
  cast_f16_kernel<<<(ndd + 255) / 256, 256, 0, stream>>>(vw, wV, ndd);
  cast_f16_kernel<<<(ndd + 255) / 256, 256, 0, stream>>>(ow, wO, ndd);
  cast_f16_kernel<<<(nff + 255) / 256, 256, 0, stream>>>(w1, w1h, nff);
  cast_f16_kernel<<<(nff + 255) / 256, 256, 0, stream>>>(w2, w2h, nff);

  // 2. LN1
  ln_kernel<<<ROWS, 256, 0, stream>>>(x, ln1_g, ln1_b, xn32, xn16);

  // 3. QKV projections (WMMA) -> head layout f16
  dim3 gD(DD / 64, ROWS / 64);
  gemm_wmma_kernel<0><<<gD, 256, 0, stream>>>(xn16, wQ, qb, nullptr, qws,
                                              ROWS, DD, DD);
  gemm_wmma_kernel<0><<<gD, 256, 0, stream>>>(xn16, wK, kb, nullptr, kws,
                                              ROWS, DD, DD);
  gemm_wmma_kernel<0><<<gD, 256, 0, stream>>>(xn16, wV, vb, nullptr, vws,
                                              ROWS, DD, DD);

  // 4. fused flash attention (+prev_attn bias), streams prev_attn once
  dim3 gA(SS / 128, BB * HH);
  attn_kernel<<<gA, 256, 0, stream>>>(qws, kws, vws, prev, scale, ctx16);

  // 5. output projection + residual (xn)
  gemm_wmma_kernel<1><<<gD, 256, 0, stream>>>(ctx16, wO, ob, xn32, xatt32,
                                              ROWS, DD, DD);

  // 6. LN2
  ln_kernel<<<ROWS, 256, 0, stream>>>(xatt32, ln2_g, ln2_b, x2_32, x2_16);

  // 7. FFN1 + exact GELU
  dim3 gF(FFN / 64, ROWS / 64);
  gemm_wmma_kernel<2><<<gF, 256, 0, stream>>>(x2_16, w1h, b1, nullptr, hid16,
                                              ROWS, FFN, DD);

  // 8. FFN2 + residual (x2) -> d_out
  gemm_wmma_kernel<3><<<gD, 256, 0, stream>>>(hid16, w2h, b2, x2_32, out,
                                              ROWS, DD, FFN);
}
